// GCN_19473381720869
// MI455X (gfx1250) — matrix operand order repaired and checked by reading, not compiled, not run on hardware
//
#include <hip/hip_runtime.h>
#include <hip/hip_bf16.h>

// ---------------------------------------------------------------------------
// GCN forward for MI455X (gfx1250, wave32, WMMA).
//   support1 = x @ W1                      (WMMA bf16 -> f32 accum)
//   agg1     = scatter_add(edge)           (global_atomic_add_f32, L2-resident)
//   h        = relu(agg1 + b1)             (fused into GEMM2 A-load)
//   support2 = h @ W2                      (WMMA)
//   out      = scatter_add(edge) + b2      (out pre-initialized with b2)
// ---------------------------------------------------------------------------

typedef __attribute__((ext_vector_type(16))) __bf16 v16bf;
typedef __attribute__((ext_vector_type(8)))  float  v8f;

#define NNODES 100000
#define NEDGES 1600000
#define NFEAT  256
#define NHID   128
#define NCLASS 32
#define MTILES (NNODES / 16)   // 6250, exact

// ---------------------------------------------------------------------------
// GEMM1: support1[100000,128] = x[100000,256] @ W1[256,128]
// Block = 256 threads = 8 waves. Whole W1 staged in LDS as bf16 WMMA-B
// fragments (64 KB): frag(kt,nt) = 32 lanes x 16 bf16 (32B contiguous/lane).
// B layout (16-bit, 32x16): lanes 0-15 -> K=kt*32+0..15, lanes 16-31 ->
// K=kt*32+16..31, N = nt*16 + lane%16.
// A layout (16-bit, 16x32): lane<16 -> K 0..15; lane>=16 -> K 16..31.
// ---------------------------------------------------------------------------
__global__ __launch_bounds__(256) void gcn_gemm1(const float* __restrict__ x,
                                                 const float* __restrict__ W1,
                                                 float* __restrict__ sup) {
  __shared__ __align__(32) __bf16 bfrag[64 * 32 * 16];  // 64 KB
  const int tid = threadIdx.x;

  // Stage + swizzle W1 (fp32 -> bf16). 2048 lane-entries, 8 per thread.
  for (int i = 0; i < 8; ++i) {
    int entry = tid * 8 + i;
    int f     = entry >> 5;          // fragment id = kt*8 + nt
    int lane  = entry & 31;
    int kt    = f >> 3;
    int nt    = f & 7;
    int n     = nt * 16 + (lane & 15);
    int kbase = kt * 32 + ((lane & 16) ? 16 : 0);
    __bf16* dst = &bfrag[(size_t)(f * 32 + lane) * 16];
#pragma unroll
    for (int k = 0; k < 16; ++k)
      dst[k] = (__bf16)W1[(size_t)(kbase + k) * NHID + n];
  }
  __syncthreads();

  const int wave  = tid >> 5;
  const int lane  = tid & 31;
  const int mtile = blockIdx.x * 8 + wave;
  if (mtile >= MTILES) return;

  const int m = mtile * 16 + (lane & 15);
  const float* __restrict__ xrow = x + (size_t)m * NFEAT;

  v8f acc[8] = {};
  for (int kt = 0; kt < 8; ++kt) {           // K = 256 in steps of 32
    const float* ap = xrow + kt * 32 + ((lane & 16) ? 16 : 0);
    v16bf a;
#pragma unroll
    for (int i = 0; i < 8; ++i) {
      a[i]     = (__bf16)ap[i];
      a[i + 8] = (__bf16)ap[8 + i];
    }
#pragma unroll
    for (int nt = 0; nt < 8; ++nt) {
      const v16bf* bp = (const v16bf*)&bfrag[(size_t)((kt * 8 + nt) * 32 + lane) * 16];
      acc[nt] = __builtin_amdgcn_wmma_f32_16x16x32_bf16(
          false, a, false, *bp, (short)0, acc[nt], false, false);
    }
  }

  // C/D layout: VGPR r -> M = r (lanes 0-15) / r+8 (lanes 16-31), N = lane%16.
  const int rbase = mtile * 16 + ((lane & 16) ? 8 : 0);
  const int col   = lane & 15;
#pragma unroll
  for (int nt = 0; nt < 8; ++nt)
#pragma unroll
    for (int r = 0; r < 8; ++r)
      sup[(size_t)(rbase + r) * NHID + nt * 16 + col] = acc[nt][r];
}

// ---------------------------------------------------------------------------
// GEMM2: support2[100000,32] = relu(agg1 + b1) @ W2[128,32]  (bias+ReLU fused)
// ---------------------------------------------------------------------------
__global__ __launch_bounds__(256) void gcn_gemm2(const float* __restrict__ agg1,
                                                 const float* __restrict__ b1,
                                                 const float* __restrict__ W2,
                                                 float* __restrict__ sup2) {
  __shared__ __align__(32) __bf16 bfrag[8 * 32 * 16];  // 8 KB (4 kt x 2 nt)
  __shared__ float b1s[NHID];
  const int tid = threadIdx.x;

  {  // one lane-entry per thread (8 frags x 32 lanes = 256)
    int f     = tid >> 5;
    int lane  = tid & 31;
    int kt    = f >> 1;
    int nt    = f & 1;
    int n     = nt * 16 + (lane & 15);
    int kbase = kt * 32 + ((lane & 16) ? 16 : 0);
    __bf16* dst = &bfrag[(size_t)(f * 32 + lane) * 16];
#pragma unroll
    for (int k = 0; k < 16; ++k)
      dst[k] = (__bf16)W2[(size_t)(kbase + k) * NCLASS + n];
  }
  if (tid < NHID) b1s[tid] = b1[tid];
  __syncthreads();

  const int wave  = tid >> 5;
  const int lane  = tid & 31;
  const int mtile = blockIdx.x * 8 + wave;
  if (mtile >= MTILES) return;

  const int m = mtile * 16 + (lane & 15);
  const float* __restrict__ arow = agg1 + (size_t)m * NHID;

  v8f acc[2] = {};
  for (int kt = 0; kt < 4; ++kt) {           // K = 128 in steps of 32
    const int kb = kt * 32 + ((lane & 16) ? 16 : 0);
    v16bf a;
#pragma unroll
    for (int i = 0; i < 8; ++i) {
      a[i]     = (__bf16)fmaxf(arow[kb + i] + b1s[kb + i], 0.0f);
      a[i + 8] = (__bf16)fmaxf(arow[kb + 8 + i] + b1s[kb + 8 + i], 0.0f);
    }
#pragma unroll
    for (int nt = 0; nt < 2; ++nt) {
      const v16bf* bp = (const v16bf*)&bfrag[(size_t)((kt * 2 + nt) * 32 + lane) * 16];
      acc[nt] = __builtin_amdgcn_wmma_f32_16x16x32_bf16(
          false, a, false, *bp, (short)0, acc[nt], false, false);
    }
  }

  const int rbase = mtile * 16 + ((lane & 16) ? 8 : 0);
  const int col   = lane & 15;
#pragma unroll
  for (int nt = 0; nt < 2; ++nt)
#pragma unroll
    for (int r = 0; r < 8; ++r)
      sup2[(size_t)(rbase + r) * NCLASS + nt * 16 + col] = acc[nt][r];
}

// ---------------------------------------------------------------------------
// SpMM scatter: agg[dst] += val * sup[src].  One thread = one float4 chunk of
// one edge. Gathers hit L2 (support fits in 192 MB L2); scatter uses
// non-returning global_atomic_add_f32 (STOREcnt path).
// ---------------------------------------------------------------------------
template <int NF4>  // feature width / 4
__global__ __launch_bounds__(256) void gcn_spmm(const float* __restrict__ sup,
                                                const int* __restrict__ esrc,
                                                const int* __restrict__ edst,
                                                const float* __restrict__ eval,
                                                float* __restrict__ agg) {
  const size_t gid = (size_t)blockIdx.x * blockDim.x + threadIdx.x;
  if (gid >= (size_t)NEDGES * NF4) return;
  const int e = (int)(gid / NF4);
  const int c = (int)(gid % NF4);

  const int   s = esrc[e];
  const int   d = edst[e];
  const float v = eval[e];

  const float4 m = ((const float4*)(sup + (size_t)s * (NF4 * 4)))[c];
  float* o = agg + (size_t)d * (NF4 * 4) + c * 4;
  __hip_atomic_fetch_add(&o[0], v * m.x, __ATOMIC_RELAXED, __HIP_MEMORY_SCOPE_AGENT);
  __hip_atomic_fetch_add(&o[1], v * m.y, __ATOMIC_RELAXED, __HIP_MEMORY_SCOPE_AGENT);
  __hip_atomic_fetch_add(&o[2], v * m.z, __ATOMIC_RELAXED, __HIP_MEMORY_SCOPE_AGENT);
  __hip_atomic_fetch_add(&o[3], v * m.w, __ATOMIC_RELAXED, __HIP_MEMORY_SCOPE_AGENT);
}

// out[i, j] = b2[j]  (bias pre-seeded so spmm2 can scatter straight into out)
__global__ __launch_bounds__(256) void gcn_init_out(const float* __restrict__ b2,
                                                    float* __restrict__ out) {
  const size_t gid = (size_t)blockIdx.x * blockDim.x + threadIdx.x;
  if (gid < (size_t)NNODES * NCLASS) out[gid] = b2[gid % NCLASS];
}

// ---------------------------------------------------------------------------
extern "C" void kernel_launch(void* const* d_in, const int* in_sizes, int n_in,
                              void* d_out, int out_size, void* d_ws, size_t ws_size,
                              hipStream_t stream) {
  (void)in_sizes; (void)n_in; (void)out_size; (void)ws_size;
  const float* x    = (const float*)d_in[0];
  const int*   esrc = (const int*)  d_in[1];
  const int*   edst = (const int*)  d_in[2];
  const float* ev   = (const float*)d_in[3];
  const float* W1   = (const float*)d_in[4];
  const float* b1   = (const float*)d_in[5];
  const float* W2   = (const float*)d_in[6];
  const float* b2   = (const float*)d_in[7];
  float* out = (float*)d_out;

  // Workspace layout (256B-aligned offsets):
  //   support1: 100000*128 f32 = 51,200,000 B
  //   agg1:     100000*128 f32 = 51,200,000 B
  //   support2: 100000*32  f32 = 12,800,000 B   (total 115.2 MB)
  char*  ws   = (char*)d_ws;
  float* sup1 = (float*)(ws);
  float* agg1 = (float*)(ws + 51200000);
  float* sup2 = (float*)(ws + 102400000);

  const int mblocks = (MTILES + 7) / 8;  // 782 blocks of 8 waves

  hipMemsetAsync(agg1, 0, (size_t)NNODES * NHID * sizeof(float), stream);
  gcn_gemm1<<<mblocks, 256, 0, stream>>>(x, W1, sup1);

  {
    const size_t total = (size_t)NEDGES * (NHID / 4);
    gcn_spmm<NHID / 4><<<(int)((total + 255) / 256), 256, 0, stream>>>(
        sup1, esrc, edst, ev, agg1);
  }

  gcn_gemm2<<<mblocks, 256, 0, stream>>>(agg1, b1, W2, sup2);

  gcn_init_out<<<(int)(((size_t)NNODES * NCLASS + 255) / 256), 256, 0, stream>>>(b2, out);

  {
    const size_t total = (size_t)NEDGES * (NCLASS / 4);
    gcn_spmm<NCLASS / 4><<<(int)((total + 255) / 256), 256, 0, stream>>>(
        sup2, esrc, edst, ev, out);
  }
}
